// VectorQuantizer_78537771974998
// MI455X (gfx1250) — matrix operand order repaired and checked by reading, not compiled, not run on hardware
//
#include <hip/hip_runtime.h>

typedef _Float16 half8 __attribute__((ext_vector_type(8)));
typedef _Float16 v16h  __attribute__((ext_vector_type(16)));
typedef float    v8f   __attribute__((ext_vector_type(8)));
typedef int      v4i   __attribute__((ext_vector_type(4)));

union V16 { v16h v; half8 h[2]; };

#define D_DIM    256
#define K_CODES  4096
#define N_ROWS   65536
#define THW      16384
#define ZQ_ELEMS 16777216
#define KT       32          // codes per LDS tile
#define NT       (K_CODES / KT)

#if defined(__has_builtin)
#  if __has_builtin(__builtin_amdgcn_global_load_async_to_lds_b128)
#    define HAVE_ASYNC_LDS 1
#  endif
#endif
#ifndef HAVE_ASYNC_LDS
#  define HAVE_ASYNC_LDS 0
#endif

#if HAVE_ASYNC_LDS
typedef __attribute__((address_space(1))) v4i as1_v4i;   // global int4
typedef __attribute__((address_space(3))) v4i as3_v4i;   // LDS int4

__device__ __forceinline__ void async_cp_b128(const void* g, void* l) {
  __builtin_amdgcn_global_load_async_to_lds_b128(
      (as1_v4i*)(void*)g, (as3_v4i*)(void*)l, 0, 0);
}
__device__ __forceinline__ void wait_async0() {
#if __has_builtin(__builtin_amdgcn_s_wait_asynccnt)
  __builtin_amdgcn_s_wait_asynccnt(0);
#else
  asm volatile("s_wait_asynccnt 0" ::: "memory");
#endif
}
#endif

// ---------------------------------------------------------------------------
// Kernel 1: embed f32 -> f16 (workspace), |e_k|^2, zero the loss accumulator.
// ---------------------------------------------------------------------------
__global__ __launch_bounds__(256) void vq_prep(const float* __restrict__ embed,
                                               _Float16* __restrict__ ef16,
                                               float* __restrict__ enorm,
                                               float* __restrict__ loss_slot) {
  int k = blockIdx.x;          // one codebook row per block
  int d = threadIdx.x;         // one channel per thread
  float v = embed[k * D_DIM + d];
  ef16[k * D_DIM + d] = (_Float16)v;
  float s = v * v;
#pragma unroll
  for (int o = 16; o > 0; o >>= 1) s += __shfl_down(s, o, 32);
  __shared__ float red[8];
  if ((threadIdx.x & 31) == 0) red[threadIdx.x >> 5] = s;
  __syncthreads();
  if (threadIdx.x == 0) {
    float t = 0.f;
#pragma unroll
    for (int i = 0; i < 8; ++i) t += red[i];
    enorm[k] = t;
    if (k == 0) *loss_slot = 0.f;   // re-zeroed on every launch (atomicAdd target)
  }
}

// ---------------------------------------------------------------------------
// Kernel 2: fused WMMA distance-GEMM + argmin + gather + loss.
// 512 blocks x 256 threads (8 waves); block owns 128 rows, loops all 4096 codes.
// Embed tiles double-buffered in LDS via GLOBAL_LOAD_ASYNC_TO_LDS (ASYNCcnt).
// ---------------------------------------------------------------------------
__global__ __launch_bounds__(256) void vq_main(const float* __restrict__ z,
                                               const _Float16* __restrict__ ef16,
                                               const float* __restrict__ enorm,
                                               const float* __restrict__ embed,
                                               float* __restrict__ zq_out,
                                               int* __restrict__ idx_out,
                                               float* __restrict__ loss_ptr) {
  // LDS: [0,32768)B  = z staging (64 rows) then embed double buffer (2 x 16KB)
  //      [32768,.. )B = per-row argmin indices (128 ints) + loss scratch (8 f32)
  __shared__ __attribute__((aligned(32))) _Float16 smem_h[16896];
  int* idxs   = (int*)(smem_h + 16384);
  float* lred = (float*)(smem_h + 16640);

  const int tid = threadIdx.x;
  const int w   = tid >> 5;          // wave 0..7
  const int L   = tid & 31;          // lane
  const int blk = blockIdx.x;
  const int bq   = blk >> 7;                 // batch index (128 blocks per batch)
  const int thwq = (blk & 127) << 7;         // thw offset of this 128-row tile

  // ---- Phase 1: stage z tile (two 64-row halves), extract A fragments ------
  // A layout (16-bit 16x32): lanes 0-15 -> M=lane, K halves {0..7,16..23};
  //                          lanes 16-31 -> M=lane-16, K halves {8..15,24..31}
  V16 afr[8];
  const int rA    = L & 15;
  const int baseA = (L < 16) ? 0 : 8;
#pragma unroll 1
  for (int half = 0; half < 2; ++half) {
    __syncthreads();
    for (int i = tid; i < 64 * D_DIM; i += 256) {
      int n = i & 63, d = i >> 6;                       // coalesced along n
      size_t ga = (size_t)(bq * D_DIM + d) * THW + thwq + (half << 6) + n;
      float v = z[ga];
      int db = d >> 3;
      smem_h[n * D_DIM + (((db ^ (n & 15)) << 3) | (d & 7))] = (_Float16)v;
    }
    __syncthreads();
    if ((w >> 2) == half) {
      int wq  = w & 3;
      int row = (wq << 4) + rA;
      const _Float16* zr = smem_h + row * D_DIM;
#pragma unroll
      for (int j = 0; j < 8; ++j) {
        int d1 = (j << 5) + baseA;
        int d2 = d1 + 16;
        afr[j].h[0] = *(const half8*)(zr + (((d1 >> 3) ^ rA) << 3));
        afr[j].h[1] = *(const half8*)(zr + (((d2 >> 3) ^ rA) << 3));
      }
    }
  }
  __syncthreads();

  // ---- Phase 2: loop over codebook in 32-code tiles, WMMA + running argmin -
  float best[8];
  int   bidx[8];
#pragma unroll
  for (int r = 0; r < 8; ++r) { best[r] = 3.4e38f; bidx[r] = 0; }

  // per-thread embed staging geometry: klocal = tid>>3, 64B (4 x b128) each
  const int    ekl  = tid >> 3;
  const int    ekk  = ekl & 15;
  const int    edb0 = (tid & 7) << 2;
  const size_t goff = (size_t)ekl * D_DIM + ((size_t)(tid & 7) << 5);

#if HAVE_ASYNC_LDS
  {   // async-stage tile 0 into buffer 0
    const _Float16* g = ef16 + goff;
    _Float16* lb = smem_h + ekl * D_DIM;
#pragma unroll
    for (int q = 0; q < 4; ++q)
      async_cp_b128(g + (q << 3), lb + (((edb0 + q) ^ ekk) << 3));
    wait_async0();
  }
  __syncthreads();
#else
  half8 rg[4];
  {   // fallback: stage tile 0 through VGPRs
    const half8* eg = (const half8*)(ef16 + goff);
#pragma unroll
    for (int q = 0; q < 4; ++q) rg[q] = eg[q];
    _Float16* ebw = smem_h + ekl * D_DIM;
#pragma unroll
    for (int q = 0; q < 4; ++q) *(half8*)(ebw + (((edb0 + q) ^ ekk) << 3)) = rg[q];
  }
  __syncthreads();
#endif

  const int offB = (L < 16) ? 0 : 8;    // B lane K-offset
#pragma unroll 1
  for (int t = 0; t < NT; ++t) {
    int cur = t & 1;

#if HAVE_ASYNC_LDS
    if (t + 1 < NT) {   // async engine fills next buffer while we do WMMAs
      const _Float16* g = ef16 + (size_t)(t + 1) * KT * D_DIM + goff;
      _Float16* lb = smem_h + ((1 - cur) << 13) + ekl * D_DIM;
#pragma unroll
      for (int q = 0; q < 4; ++q)
        async_cp_b128(g + (q << 3), lb + (((edb0 + q) ^ ekk) << 3));
    }
#else
    if (t + 1 < NT) {   // global -> regs for next tile (overlaps with WMMA)
      const half8* eg = (const half8*)(ef16 + (size_t)(t + 1) * KT * D_DIM + goff);
#pragma unroll
      for (int q = 0; q < 4; ++q) rg[q] = eg[q];
    }
#endif

    const _Float16* eb = smem_h + (cur << 13);   // 8192 halves per buffer
#pragma unroll
    for (int c = 0; c < 2; ++c) {                // two 16-code subtiles
      v8f acc = {0.f, 0.f, 0.f, 0.f, 0.f, 0.f, 0.f, 0.f};
      int klocal = (c << 4) + (L & 15);          // B: lane -> column N
      int kk = klocal & 15;
      const _Float16* er = eb + klocal * D_DIM;
#pragma unroll
      for (int j = 0; j < 8; ++j) {              // D=256 in 8 k-steps of 32
        V16 bf;
        int d0 = (j << 5) + offB;
        bf.h[0] = *(const half8*)(er + ((((d0 >> 3)    ) ^ kk) << 3));
        bf.h[1] = *(const half8*)(er + ((((d0 >> 3) + 2) ^ kk) << 3));
        acc = __builtin_amdgcn_wmma_f32_16x16x32_f16(
            false, afr[j].v, false, bf.v, (short)0, acc, false, false);
      }
      int kg = (t << 5) + (c << 4) + (L & 15);
      float en = enorm[kg];
#pragma unroll
      for (int r = 0; r < 8; ++r) {              // score = |e|^2 - 2 z.e
        float s = en - 2.0f * acc[r];
        if (s < best[r]) { best[r] = s; bidx[r] = kg; }
      }
    }

#if HAVE_ASYNC_LDS
    if (t + 1 < NT) wait_async0();   // my wave's async fills are in LDS
#else
    if (t + 1 < NT) {   // regs -> other LDS buffer
      _Float16* ebw = smem_h + ((1 - cur) << 13) + ekl * D_DIM;
#pragma unroll
      for (int q = 0; q < 4; ++q) *(half8*)(ebw + (((edb0 + q) ^ ekk) << 3)) = rg[q];
    }
#endif
    __syncthreads();
  }

  // ---- Phase 3: cross-lane argmin (16 lanes hold the 16 N-positions) -------
#pragma unroll
  for (int r = 0; r < 8; ++r) {
#pragma unroll
    for (int m = 1; m <= 8; m <<= 1) {
      float ob = __shfl_xor(best[r], m, 32);
      int   oi = __shfl_xor(bidx[r], m, 32);
      if (ob < best[r] || (ob == best[r] && oi < bidx[r])) { best[r] = ob; bidx[r] = oi; }
    }
  }
  if ((L & 15) == 0) {
    int mbase = (L >> 4) << 3;   // lanes 0-15 -> M 0..7, lanes 16-31 -> M 8..15
#pragma unroll
    for (int r = 0; r < 8; ++r) idxs[(w << 4) + mbase + r] = bidx[r];
  }
  __syncthreads();

  // ---- Phase 4: indices out, gather z_q (f32 embed), fused loss ------------
  if (tid < 128) idx_out[blk * 128 + tid] = idxs[tid];

  float lsum = 0.f;
  for (int i = tid; i < 128 * D_DIM; i += 256) {
    int n = i & 127, d = i >> 7;                 // coalesced writes along n
    int k = idxs[n];
    float e = embed[k * D_DIM + d];
    size_t a = (size_t)(bq * D_DIM + d) * THW + thwq + n;
    float df = e - z[a];
    lsum += df * df;
    zq_out[a] = e;
  }
#pragma unroll
  for (int o = 16; o > 0; o >>= 1) lsum += __shfl_down(lsum, o, 32);
  if (L == 0) lred[w] = lsum;
  __syncthreads();
  if (tid == 0) {
    float s = 0.f;
#pragma unroll
    for (int i = 0; i < 8; ++i) s += lred[i];
    // vq_loss = (1 + BETA) * sum / (4*256*16*32*32)
    atomicAdd(loss_ptr, s * (1.25f / 16777216.0f));
  }
}

// ---------------------------------------------------------------------------
extern "C" void kernel_launch(void* const* d_in, const int* in_sizes, int n_in,
                              void* d_out, int out_size, void* d_ws, size_t ws_size,
                              hipStream_t stream) {
  const float* z     = (const float*)d_in[0];   // [4,256,16,32,32] f32
  const float* embed = (const float*)d_in[1];   // [4096,256] f32

  float* out      = (float*)d_out;
  float* loss_ptr = out + ZQ_ELEMS;                 // scalar after z_q_st
  int*   idx_out  = (int*)d_out + ZQ_ELEMS + 1;     // [65536] indices after loss

  _Float16* ef16  = (_Float16*)d_ws;                            // 2 MB
  float*    enorm = (float*)((char*)d_ws + (size_t)K_CODES * D_DIM * 2); // 16 KB

  vq_prep<<<K_CODES, 256, 0, stream>>>(embed, ef16, enorm, loss_ptr);
  vq_main<<<N_ROWS / 128, 256, 0, stream>>>(z, ef16, enorm, embed,
                                            out, idx_out, loss_ptr);
}
